// GQAAttention_59923383714637
// MI455X (gfx1250) — compile-verified, matrix operand order checked by reading, not run on hardware
//
#include <hip/hip_runtime.h>

// ---------------------------------------------------------------------------
// GQA attention forward for MI455X (gfx1250), bf16 WMMA with f32 accumulation.
// Reference: q/k/v proj -> RoPE -> causal GQA softmax attention -> out proj.
// Outputs (concatenated in d_out, all f32): out [B,S,DIM], k [B,NKV,S,HD],
// v [B,NKV,S,HD].
// ---------------------------------------------------------------------------

typedef __bf16 bf16_t;
typedef __attribute__((ext_vector_type(16))) __bf16 v16bf;
typedef __attribute__((ext_vector_type(8)))  float  v8f;

#define B_    2
#define S_    2048
#define DIM_  4096
#define NH_   32
#define NKV_  8
#define HD_   128
#define NREP_ 4
#define SCALE_ 0.08838834764831845f   // 1/sqrt(128)
#define NEGINF_ (-1.0e30f)

static __device__ __forceinline__ v8f wmma_bf16(v16bf a, v16bf b, v8f c) {
  return __builtin_amdgcn_wmma_f32_16x16x32_bf16(false, a, false, b, (short)0, c,
                                                 false, false);
}

// Load 16 bf16 for one lane of an A/B operand of wmma_f32_16x16x32_bf16.
// Lane layout (16-bit operands, K=32): lanes 0-15 hold K = base..base+7 and
// base+16..base+23 with base=0; lanes 16-31 the same with base=8.
// rowp points at the start of this lane's row (M row for A, N column's
// K-major row for B); elements are contiguous over K in memory.
static __device__ __forceinline__ v16bf ld_bf16_k32(const bf16_t* rowp, int koff,
                                                    int base) {
  const bf16_t* p0 = rowp + koff + base;
  const bf16_t* p1 = p0 + 16;
  v16bf r;
#pragma unroll
  for (int i = 0; i < 8; ++i) {
    r[i]     = p0[i];
    r[8 + i] = p1[i];
  }
  return r;
}

// ---------------------------------------------------------------------------
// Generic GEMM: C[M,N] (f32) = A[M,K] @ W[K,N], A is f32 or bf16 row-major,
// W is f32 row-major (converted to bf16 on the fly). One wave computes a
// 64(M) x 16(N) tile: B tile loaded once per K-chunk, reused by 4 A tiles.
// ---------------------------------------------------------------------------
template <bool ABF16>
__global__ void __launch_bounds__(128, 1)
gqa_gemm_kernel(const void* __restrict__ Av,
                const float* __restrict__ W,
                float* __restrict__ C, int M, int N, int K) {
  const int lane = threadIdx.x & 31;
  const int wid  = threadIdx.x >> 5;
  const int wlin = blockIdx.x * 4 + wid;
  const int ntiles = N >> 4;
  const int mt = wlin / ntiles;
  const int nt = wlin - mt * ntiles;
  const int m0 = mt * 64;
  const int n0 = nt * 16;
  if (m0 >= M) return;

  const int r    = lane & 15;
  const int base = (lane >> 4) * 8;

  v8f acc[4] = {v8f{}, v8f{}, v8f{}, v8f{}};

  for (int k0 = 0; k0 < K; k0 += 32) {
    // B tile: lane owns column n0+r, needs W[k][n0+r] for its 16 k's.
    v16bf bb;
    const float* wp = W + n0 + r;
#pragma unroll
    for (int i = 0; i < 8; ++i) {
      bb[i]     = (bf16_t)wp[(long)(k0 + base + i) * N];
      bb[8 + i] = (bf16_t)wp[(long)(k0 + 16 + base + i) * N];
    }
#pragma unroll
    for (int mi = 0; mi < 4; ++mi) {
      const int row = m0 + mi * 16 + r;
      v16bf aa;
      if constexpr (ABF16) {
        const bf16_t* ap = (const bf16_t*)Av + (long)row * K;
        aa = ld_bf16_k32(ap, k0, base);
      } else {
        const float* ap = (const float*)Av + (long)row * K + k0;
#pragma unroll
        for (int i = 0; i < 8; ++i) {
          aa[i]     = (bf16_t)ap[base + i];
          aa[8 + i] = (bf16_t)ap[16 + base + i];
        }
      }
      acc[mi] = wmma_bf16(aa, bb, acc[mi]);
    }
  }

  // C layout: VGPR e holds row (e + 8*(lane>=16)), lane%16 is the column.
  const int half = lane >> 4;
#pragma unroll
  for (int mi = 0; mi < 4; ++mi) {
#pragma unroll
    for (int e = 0; e < 8; ++e) {
      C[(long)(m0 + mi * 16 + e + 8 * half) * N + n0 + r] = acc[mi][e];
    }
  }
}

// ---------------------------------------------------------------------------
// RoPE on q (f32 in, bf16 out, relayout [b,s,h,hd] -> [b,h,s,hd])
// ---------------------------------------------------------------------------
__global__ void gqa_rope_q_kernel(const float* __restrict__ qf,
                                  const float* __restrict__ cosp,
                                  const float* __restrict__ sinp,
                                  bf16_t* __restrict__ qb) {
  const int idx = blockIdx.x * blockDim.x + threadIdx.x;  // 2^23 threads
  const int d = idx & 63;
  const int h = (idx >> 6) & 31;
  const int s = (idx >> 11) & 2047;
  const int b = idx >> 22;
  const float* src = qf + (((long)(b * S_ + s) * NH_ + h) << 7) + d;
  const float x0 = src[0], x1 = src[64];
  const float c = cosp[s * HD_ + d], sn = sinp[s * HD_ + d];
  const float y0 = x0 * c - x1 * sn;   // cos/sin duplicated over the two halves
  const float y1 = x1 * c + x0 * sn;
  bf16_t* dst = qb + (((long)(b * NH_ + h) * S_ + s) << 7) + d;
  dst[0]  = (bf16_t)y0;
  dst[64] = (bf16_t)y1;
}

// RoPE on k; emits bf16 [b,kvh,s,hd] for attention and f32 k output.
__global__ void gqa_rope_k_kernel(const float* __restrict__ kf,
                                  const float* __restrict__ cosp,
                                  const float* __restrict__ sinp,
                                  bf16_t* __restrict__ kb,
                                  float* __restrict__ kout) {
  const int idx = blockIdx.x * blockDim.x + threadIdx.x;  // 2^21 threads
  const int d = idx & 63;
  const int h = (idx >> 6) & 7;
  const int s = (idx >> 9) & 2047;
  const int b = idx >> 20;
  const float* src = kf + (((long)(b * S_ + s) * NKV_ + h) << 7) + d;
  const float x0 = src[0], x1 = src[64];
  const float c = cosp[s * HD_ + d], sn = sinp[s * HD_ + d];
  const float y0 = x0 * c - x1 * sn;
  const float y1 = x1 * c + x0 * sn;
  const long o = (((long)(b * NKV_ + h) * S_ + s) << 7) + d;
  kb[o]        = (bf16_t)y0;
  kb[o + 64]   = (bf16_t)y1;
  kout[o]      = y0;
  kout[o + 64] = y1;
}

// v: relayout to f32 [b,kvh,s,hd] output and bf16 transposed [b,kvh,hd,s].
__global__ void gqa_v_reformat_kernel(const float* __restrict__ vf,
                                      float* __restrict__ vout,
                                      bf16_t* __restrict__ vbt) {
  const int idx = blockIdx.x * blockDim.x + threadIdx.x;  // 2^22 threads
  const int d = idx & 127;
  const int h = (idx >> 7) & 7;
  const int s = (idx >> 10) & 2047;
  const int b = idx >> 21;
  const float val = vf[((long)(b * S_ + s) * NKV_ + h) * HD_ + d];
  vout[((long)(b * NKV_ + h) * S_ + s) * HD_ + d] = val;
  vbt[((long)(b * NKV_ + h) * HD_ + d) * (long)S_ + s] = (bf16_t)val;
}

// ---------------------------------------------------------------------------
// Flash attention: one wave per (b, head, 16-row q tile). 32 keys/iteration,
// online softmax in f32, causal mask analytic, P transposed via wave-private
// LDS, PV contraction uses the transposed-V buffer for contiguous B loads.
// __launch_bounds__(128,1): optimize for 1 wave/SIMD so Q tiles + 8 f32
// accumulator tiles (~170 VGPRs) stay fully register-resident (no spills).
// ---------------------------------------------------------------------------
__global__ void __launch_bounds__(128, 1)
gqa_flash_kernel(const bf16_t* __restrict__ qb,
                 const bf16_t* __restrict__ kb,
                 const bf16_t* __restrict__ vbt,
                 bf16_t* __restrict__ ob) {
  __shared__ float plds[4][16][33];
  const int lane = threadIdx.x & 31;
  const int wid  = threadIdx.x >> 5;
  const int wlin = blockIdx.x * 4 + wid;   // 8192 waves total
  const int qt = wlin & 127;               // S/16
  const int h  = (wlin >> 7) & 31;
  const int b  = wlin >> 12;
  const int kvh = h >> 2;                  // NREP = 4
  const int q0 = qt * 16;
  const int r    = lane & 15;
  const int half = lane >> 4;
  const int base = half * 8;

  // Resident Q tile (A-layout), 4 chunks of K=32 over HD=128.
  const bf16_t* qrow = qb + (((long)(b * NH_ + h) * S_ + q0 + r) << 7);
  v16bf qa[4];
#pragma unroll
  for (int c = 0; c < 4; ++c) qa[c] = ld_bf16_k32(qrow, c * 32, base);

  v8f o[8];
#pragma unroll
  for (int t = 0; t < 8; ++t) o[t] = v8f{};
  float m_[8], l_[8];
#pragma unroll
  for (int e = 0; e < 8; ++e) { m_[e] = NEGINF_; l_[e] = 0.0f; }

  const bf16_t* kbp = kb + (((long)(b * NKV_ + kvh) * S_) << 7);
  const bf16_t* vbp = vbt + ((long)(b * NKV_ + kvh) * HD_) * (long)S_;
  float (*P)[33] = plds[wid];

  const int nblk = (q0 + 47) >> 5;   // keys 0 .. q0+15 inclusive
  for (int jb = 0; jb < nblk; ++jb) {
    const int j0 = jb * 32;
    // S = Q K^T for 32 keys (two 16x16 C tiles).
    v8f s0 = v8f{}, s1 = v8f{};
    const bf16_t* k0p = kbp + ((long)(j0 + r) << 7);
    const bf16_t* k1p = kbp + ((long)(j0 + 16 + r) << 7);
#pragma unroll
    for (int c = 0; c < 4; ++c) {
      v16bf b0 = ld_bf16_k32(k0p, c * 32, base);
      s0 = wmma_bf16(qa[c], b0, s0);
      v16bf b1 = ld_bf16_k32(k1p, c * 32, base);
      s1 = wmma_bf16(qa[c], b1, s1);
    }
    // Online softmax. Row (e + 8*half) lives in lanes of this half only, so
    // xor-butterflies with masks <16 reduce exactly over the row.
    const int col0 = j0 + r, col1 = col0 + 16;
    float p0[8], p1[8], alpha[8];
#pragma unroll
    for (int e = 0; e < 8; ++e) {
      const int row = q0 + e + 8 * half;
      const float v0 = (col0 <= row) ? s0[e] * SCALE_ : NEGINF_;
      const float v1 = (col1 <= row) ? s1[e] * SCALE_ : NEGINF_;
      float mx = fmaxf(v0, v1);
      mx = fmaxf(mx, __shfl_xor(mx, 1, 32));
      mx = fmaxf(mx, __shfl_xor(mx, 2, 32));
      mx = fmaxf(mx, __shfl_xor(mx, 4, 32));
      mx = fmaxf(mx, __shfl_xor(mx, 8, 32));
      const float mnew = fmaxf(m_[e], mx);
      alpha[e] = __expf(m_[e] - mnew);
      m_[e] = mnew;
      p0[e] = __expf(v0 - mnew);
      p1[e] = __expf(v1 - mnew);
      float sum = p0[e] + p1[e];
      sum += __shfl_xor(sum, 1, 32);
      sum += __shfl_xor(sum, 2, 32);
      sum += __shfl_xor(sum, 4, 32);
      sum += __shfl_xor(sum, 8, 32);
      l_[e] = l_[e] * alpha[e] + sum;
    }
#pragma unroll
    for (int t = 0; t < 8; ++t)
#pragma unroll
      for (int e = 0; e < 8; ++e) o[t][e] *= alpha[e];

    // Transpose P (C layout) -> A layout through wave-private LDS.
#pragma unroll
    for (int e = 0; e < 8; ++e) {
      P[e + 8 * half][r]      = p0[e];
      P[e + 8 * half][16 + r] = p1[e];
    }
    v16bf pa;
#pragma unroll
    for (int i = 0; i < 8; ++i) {
      pa[i]     = (bf16_t)P[r][base + i];
      pa[8 + i] = (bf16_t)P[r][16 + base + i];
    }
    // O += P V ; V^T buffer makes B loads contiguous over the 32 keys.
#pragma unroll
    for (int nc = 0; nc < 8; ++nc) {
      const bf16_t* vrow = vbp + (long)(nc * 16 + r) * (long)S_;
      v16bf vb = ld_bf16_k32(vrow, j0, base);
      o[nc] = wmma_bf16(pa, vb, o[nc]);
    }
  }

  // Normalize and write attn output as bf16 [b, s, h*HD + d].
#pragma unroll
  for (int e = 0; e < 8; ++e) {
    const float inv = 1.0f / l_[e];
    const long rowoff = ((long)b * S_ + q0 + e + 8 * half) * (long)(NH_ * HD_) +
                        h * HD_ + r;
#pragma unroll
    for (int nc = 0; nc < 8; ++nc) {
      ob[rowoff + nc * 16] = (bf16_t)(o[nc][e] * inv);
    }
  }
}

// ---------------------------------------------------------------------------
extern "C" void kernel_launch(void* const* d_in, const int* in_sizes, int n_in,
                              void* d_out, int out_size, void* d_ws, size_t ws_size,
                              hipStream_t stream) {
  const float* x    = (const float*)d_in[0];
  const float* cosp = (const float*)d_in[1];
  const float* sinp = (const float*)d_in[2];
  // d_in[3] = mask (causal mask computed analytically in-kernel)
  const float* wq = (const float*)d_in[4];
  const float* wk = (const float*)d_in[5];
  const float* wv = (const float*)d_in[6];
  const float* wo = (const float*)d_in[7];

  float* out_f = (float*)d_out;                                  // [B,S,DIM]
  float* k_out = out_f + (long)B_ * S_ * DIM_;                   // [B,NKV,S,HD]
  float* v_out = k_out + (long)B_ * NKV_ * S_ * HD_;             // [B,NKV,S,HD]

  char* ws = (char*)d_ws;
  float*  q_f32   = (float*)ws;                 ws += (size_t)4096 * 4096 * 4;  // [b,s,h,hd]
  float*  k_f32   = (float*)ws;                 ws += (size_t)4096 * 1024 * 4;
  float*  v_f32   = (float*)ws;                 ws += (size_t)4096 * 1024 * 4;
  bf16_t* q_bf    = (bf16_t*)ws;                ws += (size_t)4096 * 4096 * 2;  // [b,h,s,hd]
  bf16_t* k_bf    = (bf16_t*)ws;                ws += (size_t)4096 * 1024 * 2;  // [b,kvh,s,hd]
  bf16_t* v_bf_t  = (bf16_t*)ws;                ws += (size_t)4096 * 1024 * 2;  // [b,kvh,hd,s]
  bf16_t* attn_bf = (bf16_t*)ws;                                                // [b,s,nh*hd]

  const int M = B_ * S_;  // 4096

  // 1) QKV projections (f32 A -> f32 C, bf16 WMMA internally).
  gqa_gemm_kernel<false><<<(M / 64) * (DIM_ / 16) / 4, 128, 0, stream>>>(
      x, wq, q_f32, M, NH_ * HD_, DIM_);
  gqa_gemm_kernel<false><<<(M / 64) * ((NKV_ * HD_) / 16) / 4, 128, 0, stream>>>(
      x, wk, k_f32, M, NKV_ * HD_, DIM_);
  gqa_gemm_kernel<false><<<(M / 64) * ((NKV_ * HD_) / 16) / 4, 128, 0, stream>>>(
      x, wv, v_f32, M, NKV_ * HD_, DIM_);

  // 2) RoPE + reformat (+ f32 k/v outputs).
  gqa_rope_q_kernel<<<(B_ * S_ * NH_ * 64) / 256, 256, 0, stream>>>(
      q_f32, cosp, sinp, q_bf);
  gqa_rope_k_kernel<<<(B_ * S_ * NKV_ * 64) / 256, 256, 0, stream>>>(
      k_f32, cosp, sinp, k_bf, k_out);
  gqa_v_reformat_kernel<<<(B_ * S_ * NKV_ * 128) / 256, 256, 0, stream>>>(
      v_f32, v_out, v_bf_t);

  // 3) Causal GQA flash attention.
  gqa_flash_kernel<<<(B_ * NH_ * (S_ / 16)) / 4, 128, 0, stream>>>(
      q_bf, k_bf, v_bf_t, attn_bf);

  // 4) Output projection (bf16 A -> f32 C).
  gqa_gemm_kernel<true><<<(M / 64) * (DIM_ / 16) / 4, 128, 0, stream>>>(
      attn_bf, wo, out_f, M, DIM_, NH_ * HD_);

  (void)in_sizes; (void)n_in; (void)out_size; (void)ws_size;
}
